// MultiheadSelfAttention_18992345383128
// MI455X (gfx1250) — matrix operand order repaired and checked by reading, not compiled, not run on hardware
//
#include <hip/hip_runtime.h>
#include <hip/hip_bf16.h>

#define BATCH 2
#define SEQ   2048
#define DM    768
#define NHEAD 12
#define DKH   64

typedef __attribute__((ext_vector_type(16))) __bf16 v16bf;
typedef __attribute__((ext_vector_type(8)))  float  v8f;

// A-fragment K index for VGPR pair v (0..7) and lane-half (0/1).
// 16-bit A 16x32: VGPR0 lanes0-15 K=0,1; lanes16-31 K=8,9; VGPR4 K=16,17 / 24,25 ...
__device__ __forceinline__ int kmap(int v, int half) {
  return ((v & 4) ? (16 + ((v & 3) << 1)) : (v << 1)) + (half << 3);
}

__device__ __forceinline__ v8f wmma_bf16(v16bf a, v16bf b, v8f c) {
  return __builtin_amdgcn_wmma_f32_16x16x32_bf16(false, a, false, b, (short)0, c,
                                                 false, false);
}

// 32-bit LDS byte offset of a shared-memory object (generic -> AS(3) cast).
__device__ __forceinline__ unsigned lds_off(const void* p) {
  return (unsigned)(size_t)(__attribute__((address_space(3))) const char*)p;
}

// Async copy 16 bytes global -> LDS (ASYNCcnt-tracked, per-wave).
__device__ __forceinline__ void async_copy_b128(unsigned lds_byte_off,
                                                const void* gptr) {
  asm volatile("global_load_async_to_lds_b128 %0, %1, off"
               :: "v"(lds_byte_off), "v"(gptr) : "memory");
}
__device__ __forceinline__ void wait_asynccnt0() {
  asm volatile("s_wait_asynccnt 0x0" ::: "memory");
}

// ---------------------------------------------------------------------------
// C = A(M x 768) @ W(768 x 768)^T, A/W fp32 -> bf16 operands, fp32 accum.
// MODE 0: C[m*768 + n] row-major
// MODE 1: per-head transposed store: C[((b*H + h)*64 + dk)*SEQ + s]
// Block: 128 threads (4 waves). Tile: 64 rows x 64 cols. K chunked by 32.
// A tile staged to LDS with global_load_async_to_lds_b128; W tile staged
// transposed as bf16 (wt[k][n]); a lane's B-fragment k values are 72 elements apart.
// ---------------------------------------------------------------------------
template <int MODE>
__global__ __launch_bounds__(128)
void gemm_xwT(const float* __restrict__ A, const float* __restrict__ W,
              float* __restrict__ C) {
  __shared__ __bf16 wt[32][72];                       // W chunk ^T, padded
  __shared__ __align__(16) float ash[4][16][36];      // per-wave A chunk, padded

  const int tid  = threadIdx.x;
  const int lane = tid & 31;
  const int wave = tid >> 5;
  const int nL   = lane & 15;
  const int half = lane >> 4;
  const int tileM = blockIdx.x * 64;
  const int tileN = blockIdx.y * 64;

  const v8f vzero = {0.f, 0.f, 0.f, 0.f, 0.f, 0.f, 0.f, 0.f};
  v8f acc[4];
#pragma unroll
  for (int s = 0; s < 4; ++s) acc[s] = vzero;

  for (int kc = 0; kc < DM; kc += 32) {
    // Async-stage this wave's A tile: rows tileM+wave*16..+15, cols kc..kc+31.
#pragma unroll
    for (int i = 0; i < 4; ++i) {
      const int s   = lane + i * 32;   // 0..127 segments of 4 floats
      const int row = s >> 3;          // 0..15
      const int kq  = (s & 7) << 2;    // 0,4,...,28
      const float* g = A + (size_t)(tileM + wave * 16 + row) * DM + kc + kq;
      async_copy_b128(lds_off(&ash[wave][row][kq]), g);
    }

    // Cooperative load of W[tileN..+63][kc..+31] -> LDS transposed as bf16.
    {
      const int n  = tid >> 1;
      const int kb = (tid & 1) << 4;
      const float* wp = W + (size_t)(tileN + n) * DM + kc + kb;
#pragma unroll
      for (int i = 0; i < 16; ++i) wt[kb + i][n] = (__bf16)wp[i];
    }
    __syncthreads();
    wait_asynccnt0();

    // A fragment from LDS: lane = row, VGPR pairs = contiguous K pair.
    v16bf af;
#pragma unroll
    for (int v = 0; v < 8; ++v) {
      const int k0 = kmap(v, half);
      float2 f = *(const float2*)&ash[wave][nL][k0];
      af[2 * v]     = (__bf16)f.x;
      af[2 * v + 1] = (__bf16)f.y;
    }

#pragma unroll
    for (int sub = 0; sub < 4; ++sub) {
      // B fragment: lane%16 = N, VGPR v = K pair kmap(v, half).
      v16bf bfr;
#pragma unroll
      for (int v = 0; v < 8; ++v) {
        const int k0 = kmap(v, half);
        bfr[2 * v]     = wt[k0][sub * 16 + nL];
        bfr[2 * v + 1] = wt[k0 + 1][sub * 16 + nL];
      }
      acc[sub] = wmma_bf16(af, bfr, acc[sub]);
    }
    __syncthreads();
  }

  // Store: C/D layout lane = N, VGPR r = row r + 8*half.
#pragma unroll
  for (int sub = 0; sub < 4; ++sub) {
#pragma unroll
    for (int r = 0; r < 8; ++r) {
      const int m = tileM + wave * 16 + r + 8 * half;
      const int n = tileN + sub * 16 + nL;
      const float val = acc[sub][r];
      if (MODE == 0) {
        C[(size_t)m * DM + n] = val;
      } else {
        const int b = m >> 11, s = m & (SEQ - 1);
        const int h = n >> 6, dk = n & 63;
        C[((size_t)(b * NHEAD + h) * DKH + dk) * SEQ + s] = val;
      }
    }
  }
}

// ---------------------------------------------------------------------------
// Flash attention forward. Q: (B,S,D) per-head cols; KT: (B,H,DK,S); V: (B,S,D).
// Block: 128 threads = 4 independent waves; wave handles one 16-row query tile,
// streams 32-key blocks with online softmax. Output O: (B,S,D) fp32.
// ---------------------------------------------------------------------------
__global__ __launch_bounds__(128)
void attn_fwd(const float* __restrict__ Q, const float* __restrict__ KT,
              const float* __restrict__ V, float* __restrict__ O) {
  __shared__ __bf16 psh[4][16][32];  // per-wave P repack buffer

  const int tid  = threadIdx.x;
  const int lane = tid & 31;
  const int wave = tid >> 5;
  const int nL   = lane & 15;
  const int half = lane >> 4;
  const int bh = blockIdx.y;
  const int b = bh / NHEAD, h = bh % NHEAD;
  const int qt = blockIdx.x * 4 + wave;  // query tile index, 0..127
  const int qBase = qt * 16;

  // Q fragments for both K chunks (DK = 64 = 2 x 32).
  v16bf qa[2];
  {
    const float* qp = Q + ((size_t)b * SEQ + qBase + nL) * DM + h * DKH;
#pragma unroll
    for (int c = 0; c < 2; ++c)
#pragma unroll
      for (int v = 0; v < 8; ++v) {
        const int k0 = c * 32 + kmap(v, half);
        float2 f = *(const float2*)(qp + k0);
        qa[c][2 * v]     = (__bf16)f.x;
        qa[c][2 * v + 1] = (__bf16)f.y;
      }
  }

  const v8f vzero = {0.f, 0.f, 0.f, 0.f, 0.f, 0.f, 0.f, 0.f};
  v8f o[4];
#pragma unroll
  for (int s = 0; s < 4; ++s) o[s] = vzero;
  float mrun[8], lrun[8];
#pragma unroll
  for (int r = 0; r < 8; ++r) { mrun[r] = -1e30f; lrun[r] = 0.f; }

  const float* ktBase = KT + (size_t)(b * NHEAD + h) * DKH * SEQ;
  const int nKB = (qt >> 1) + 1;  // causal: 32-key blocks needed

  for (int jb = 0; jb < nKB; ++jb) {
    const int jBase = jb * 32;

    // Scores: S = Q @ K^T for 32 keys (two 16-col D tiles).
    v8f sLo = vzero, sHi = vzero;
#pragma unroll
    for (int c = 0; c < 2; ++c) {
      // B-frag from kT: lane%16 = key, VGPR v = feature pair kmap(v, half).
      const float* kp = ktBase + (size_t)(c * 32) * SEQ + jBase + nL;
      v16bf bLo, bHi;
#pragma unroll
      for (int v = 0; v < 8; ++v) {
        const float* r0 = kp + (size_t)kmap(v, half) * SEQ;
        bLo[2 * v] = (__bf16)r0[0]; bLo[2 * v + 1] = (__bf16)r0[SEQ];
        bHi[2 * v] = (__bf16)r0[16]; bHi[2 * v + 1] = (__bf16)r0[SEQ + 16];
      }
      sLo = wmma_bf16(qa[c], bLo, sLo);
      sHi = wmma_bf16(qa[c], bHi, sHi);
    }

    // Scale + causal mask + online softmax (row = r + 8*half, col = lane%16).
    float pLo[8], pHi[8];
#pragma unroll
    for (int r = 0; r < 8; ++r) {
      const int qi = qBase + r + 8 * half;
      float sl = sLo[r] * 0.125f;  // rsqrt(64)
      float sh = sHi[r] * 0.125f;
      if (jBase + nL > qi) sl = -1e30f;
      if (jBase + 16 + nL > qi) sh = -1e30f;
      float rm = fmaxf(sl, sh);
#pragma unroll
      for (int x = 8; x >= 1; x >>= 1) rm = fmaxf(rm, __shfl_xor(rm, x, 32));
      const float mNew  = fmaxf(mrun[r], rm);
      const float alpha = __expf(mrun[r] - mNew);
      const float el = __expf(sl - mNew);
      const float eh = __expf(sh - mNew);
      float rs = el + eh;
#pragma unroll
      for (int x = 8; x >= 1; x >>= 1) rs += __shfl_xor(rs, x, 32);
      lrun[r] = lrun[r] * alpha + rs;
      mrun[r] = mNew;
      pLo[r] = el;
      pHi[r] = eh;
#pragma unroll
      for (int sub = 0; sub < 4; ++sub) o[sub][r] *= alpha;
    }

    // Repack P from C-layout to A-layout via per-wave LDS (DS pipe is in-order
    // within a wave; wave32 lockstep -> no barrier needed).
#pragma unroll
    for (int r = 0; r < 8; ++r) {
      const int m = r + 8 * half;
      psh[wave][m][nL]      = (__bf16)pLo[r];
      psh[wave][m][16 + nL] = (__bf16)pHi[r];
    }
    v16bf pa;
#pragma unroll
    for (int v = 0; v < 8; ++v) {
      const int k0 = kmap(v, half);
      pa[2 * v]     = psh[wave][nL][k0];
      pa[2 * v + 1] = psh[wave][nL][k0 + 1];
    }

    // O += P @ V ; B-frag from V natural layout: lane%16 = feat, VGPR = key pair kmap(v, half).
    const float* vp = V + ((size_t)b * SEQ + jBase) * DM + h * DKH + nL;
#pragma unroll
    for (int sub = 0; sub < 4; ++sub) {
      v16bf vb;
#pragma unroll
      for (int v = 0; v < 8; ++v) {
        const int k0 = kmap(v, half);
        vb[2 * v]     = (__bf16)vp[(size_t)k0 * DM + sub * 16];
        vb[2 * v + 1] = (__bf16)vp[(size_t)(k0 + 1) * DM + sub * 16];
      }
      o[sub] = wmma_bf16(pa, vb, o[sub]);
    }
  }

  // Normalize and store (B,S,D) fp32.
#pragma unroll
  for (int r = 0; r < 8; ++r) {
    const float inv = 1.0f / lrun[r];
    const int m = qBase + r + 8 * half;
    float* op = O + ((size_t)b * SEQ + m) * DM + h * DKH + nL;
#pragma unroll
    for (int sub = 0; sub < 4; ++sub) op[sub * 16] = o[sub][r] * inv;
  }
}

// ---------------------------------------------------------------------------
extern "C" void kernel_launch(void* const* d_in, const int* in_sizes, int n_in,
                              void* d_out, int out_size, void* d_ws,
                              size_t ws_size, hipStream_t stream) {
  const float* x  = (const float*)d_in[0];
  const float* Wq = (const float*)d_in[1];
  const float* Wk = (const float*)d_in[2];
  const float* Wv = (const float*)d_in[3];
  const float* Wo = (const float*)d_in[4];

  const size_t elems = (size_t)BATCH * SEQ * DM;  // 3.1M floats each
  float* q  = (float*)d_ws;
  float* kT = q + elems;
  float* v  = kT + elems;
  float* ao = v + elems;

  dim3 gemmGrid(/*M/64*/ (BATCH * SEQ) / 64, /*N/64*/ DM / 64);
  dim3 gemmBlk(128);

  gemm_xwT<0><<<gemmGrid, gemmBlk, 0, stream>>>(x, Wq, q);    // Q normal
  gemm_xwT<1><<<gemmGrid, gemmBlk, 0, stream>>>(x, Wk, kT);   // K transposed/head
  gemm_xwT<0><<<gemmGrid, gemmBlk, 0, stream>>>(x, Wv, v);    // V normal

  dim3 attnGrid(SEQ / 16 / 4, BATCH * NHEAD);  // (32, 24)
  attn_fwd<<<attnGrid, dim3(128), 0, stream>>>(q, kT, v, ao);

  gemm_xwT<0><<<gemmGrid, gemmBlk, 0, stream>>>(ao, Wo, (float*)d_out);
}
